// SCNLayer_27711128994333
// MI455X (gfx1250) — matrix operand order repaired and checked by reading, not compiled, not run on hardware
//
#include <hip/hip_runtime.h>
#include <stdint.h>

// ---------------------------------------------------------------------------
// SCN layer for MI455X (gfx1250):
//   stage1: AGG = L @ X + X          (split-bf16 WMMA, 3 gemms for f32 accuracy)
//   stage2: OUT = relu(AGG @ W^T + b) (in-place on d_out, row-exclusive per wave)
// ---------------------------------------------------------------------------

typedef __bf16 bf16_t;
typedef __attribute__((ext_vector_type(8)))  bf16_t          v8bf;
typedef __attribute__((ext_vector_type(16))) bf16_t          v16bf;
typedef __attribute__((ext_vector_type(8)))  float           v8f;
typedef __attribute__((ext_vector_type(8)))  unsigned short  v8u;
typedef __attribute__((ext_vector_type(4)))  unsigned short  v4u;

#define D_DIM 128   // feature dim
#define BM    128   // rows per block (8 waves x 16)
#define KB    32    // K-chunk (one bf16 WMMA K)
#define LDST  40    // LDS row stride in ushorts (80B: 16B-aligned, bank-spread)

// Split f32 v into hi (truncated bf16, remainder exact) and lo (RNE bf16 of rem).
__device__ __forceinline__ void split1(float v, unsigned short& h, unsigned short& l) {
  unsigned u = __float_as_uint(v);
  h = (unsigned short)(u >> 16);
  float rem = v - __uint_as_float(u & 0xffff0000u);      // exact
  unsigned ur = __float_as_uint(rem);
  l = (unsigned short)((ur + 0x7fffu + ((ur >> 16) & 1u)) >> 16);  // RNE to bf16
}

__device__ __forceinline__ v16bf combine16(v8u a, v8u b) {
  v8bf ab = __builtin_bit_cast(v8bf, a);
  v8bf bb = __builtin_bit_cast(v8bf, b);
  return __builtin_shufflevector(ab, bb, 0,1,2,3,4,5,6,7,8,9,10,11,12,13,14,15);
}

// Load 16 bf16 (as two 16B halves) from LDS ushort array.
__device__ __forceinline__ v16bf ld2x8(const unsigned short* s, int off0, int off1) {
  v8u a = *(const v8u*)(s + off0);
  v8u b = *(const v8u*)(s + off1);
  return combine16(a, b);
}

// Convert 8 consecutive global f32 into hi/lo bf16 octets.
__device__ __forceinline__ void cvt8(const float* __restrict__ p, v8u& h, v8u& l) {
  float4 f0 = *(const float4*)p;
  float4 f1 = *(const float4*)(p + 4);
  unsigned short hh, ll;
  split1(f0.x, hh, ll); h[0]=hh; l[0]=ll;
  split1(f0.y, hh, ll); h[1]=hh; l[1]=ll;
  split1(f0.z, hh, ll); h[2]=hh; l[2]=ll;
  split1(f0.w, hh, ll); h[3]=hh; l[3]=ll;
  split1(f1.x, hh, ll); h[4]=hh; l[4]=ll;
  split1(f1.y, hh, ll); h[5]=hh; l[5]=ll;
  split1(f1.z, hh, ll); h[6]=hh; l[6]=ll;
  split1(f1.w, hh, ll); h[7]=hh; l[7]=ll;
}

#define WMMA_BF16(A, B, C) \
  __builtin_amdgcn_wmma_f32_16x16x32_bf16(false, (A), false, (B), (short)0, (C), false, false)

// ---------------------------------------------------------------------------
// Stage 1: AGG = L @ X + X  (per branch via blockIdx.y), AGG written to `out`.
// ---------------------------------------------------------------------------
__global__ __launch_bounds__(256) void scn_stage1(
    const float* __restrict__ X0, const float* __restrict__ X1, const float* __restrict__ X2,
    const float* __restrict__ L0, const float* __restrict__ L1, const float* __restrict__ L2,
    float* __restrict__ out, int N0_, int N1_, int N2_)
{
  __shared__ unsigned short sLhi[BM * LDST];
  __shared__ unsigned short sLlo[BM * LDST];
  __shared__ unsigned short sXhi[D_DIM * LDST];   // transposed: [n][k]
  __shared__ unsigned short sXlo[D_DIM * LDST];

  const int br = blockIdx.y;
  const float* L; const float* X; float* agg; int n;
  if (br == 0)      { L = L0; X = X0; agg = out;                              n = N0_; }
  else if (br == 1) { L = L1; X = X1; agg = out + (size_t)N0_ * D_DIM;        n = N1_; }
  else              { L = L2; X = X2; agg = out + (size_t)(N0_ + N1_) * D_DIM; n = N2_; }

  const int mblk = blockIdx.x * BM;
  if (mblk >= n) return;

  const int tid  = threadIdx.x;
  const int lane = tid & 31;
  const int wave = tid >> 5;          // 0..7
  const int l16  = lane & 15;
  const int half = lane >> 4;         // 0 or 1

  v8f zero = {0.f, 0.f, 0.f, 0.f, 0.f, 0.f, 0.f, 0.f};
  v8f acc[8];
  #pragma unroll
  for (int t = 0; t < 8; ++t) acc[t] = zero;

  // cooperative-load decompositions
  const int lr  = tid >> 1;           // L tile row 0..127
  const int lc0 = (tid & 1) * 16;     // L tile col base {0,16}
  const int xk  = tid >> 3;           // X tile k row 0..31
  const int xn0 = (tid & 7) * 16;     // X tile col base 0..112

  for (int k0 = 0; k0 < n; k0 += KB) {
    // ---- stage L tile (BM x KB) into LDS as hi/lo bf16 ----
    const float* Lp = L + (size_t)(mblk + lr) * n + (k0 + lc0);
    #pragma unroll
    for (int i = 0; i < 4; ++i) {
      float4 f = *(const float4*)(Lp + 4 * i);
      v4u h, l; unsigned short hh, ll;
      split1(f.x, hh, ll); h[0]=hh; l[0]=ll;
      split1(f.y, hh, ll); h[1]=hh; l[1]=ll;
      split1(f.z, hh, ll); h[2]=hh; l[2]=ll;
      split1(f.w, hh, ll); h[3]=hh; l[3]=ll;
      *(v4u*)&sLhi[lr * LDST + lc0 + 4 * i] = h;
      *(v4u*)&sLlo[lr * LDST + lc0 + 4 * i] = l;
    }
    // ---- stage X chunk (KB x D) transposed into LDS as hi/lo bf16 ----
    const float* Xp = X + (size_t)(k0 + xk) * D_DIM + xn0;
    #pragma unroll
    for (int i = 0; i < 4; ++i) {
      float4 f = *(const float4*)(Xp + 4 * i);
      unsigned short hh, ll;
      split1(f.x, hh, ll); sXhi[(xn0 + 4*i + 0) * LDST + xk] = hh; sXlo[(xn0 + 4*i + 0) * LDST + xk] = ll;
      split1(f.y, hh, ll); sXhi[(xn0 + 4*i + 1) * LDST + xk] = hh; sXlo[(xn0 + 4*i + 1) * LDST + xk] = ll;
      split1(f.z, hh, ll); sXhi[(xn0 + 4*i + 2) * LDST + xk] = hh; sXlo[(xn0 + 4*i + 2) * LDST + xk] = ll;
      split1(f.w, hh, ll); sXhi[(xn0 + 4*i + 3) * LDST + xk] = hh; sXlo[(xn0 + 4*i + 3) * LDST + xk] = ll;
    }
    if (k0 + KB < n) {                       // pull next chunk toward L2/L0
      __builtin_prefetch(Lp + KB, 0, 1);
      __builtin_prefetch(Xp + (size_t)KB * D_DIM, 0, 1);
    }
    __syncthreads();

    // ---- A fragments for this wave's 16 rows ----
    // 16-bit A 16x32 layout: lanes 0-15 -> K {0..7,16..23}, lanes 16-31 -> K {8..15,24..31}
    const int arow = wave * 16 + l16;
    const int akk  = half * 8;
    v16bf Ahi = ld2x8(sLhi, arow * LDST + akk, arow * LDST + akk + 16);
    v16bf Alo = ld2x8(sLlo, arow * LDST + akk, arow * LDST + akk + 16);

    #pragma unroll
    for (int t = 0; t < 8; ++t) {
      // B 32x16 layout: lanes 0-15 -> K {0..7,16..23}, lanes 16-31 -> K {8..15,24..31}; N = lane%16
      const int boff = (t * 16 + l16) * LDST + half * 8;
      v16bf Bhi = ld2x8(sXhi, boff, boff + 16);
      v16bf Blo = ld2x8(sXlo, boff, boff + 16);
      acc[t] = WMMA_BF16(Ahi, Bhi, acc[t]);
      acc[t] = WMMA_BF16(Ahi, Blo, acc[t]);
      acc[t] = WMMA_BF16(Alo, Bhi, acc[t]);
    }
    __syncthreads();
  }

  // ---- epilogue: AGG = acc + X, store to out (row-major [n][128]) ----
  // C/D layout: VGPR v -> M = v + 8*half, N = 16t + l16
  #pragma unroll
  for (int t = 0; t < 8; ++t) {
    const int nn = t * 16 + l16;
    #pragma unroll
    for (int v = 0; v < 8; ++v) {
      const int m = mblk + wave * 16 + v + 8 * half;
      agg[(size_t)m * D_DIM + nn] = acc[t][v] + X[(size_t)m * D_DIM + nn];
    }
  }
}

// ---------------------------------------------------------------------------
// Stage 2: OUT = relu(AGG @ W^T + b), in place on `out`.
// Each wave reads only its own 16 rows; all reads precede its stores.
// ---------------------------------------------------------------------------
__global__ __launch_bounds__(256) void scn_stage2(
    const float* __restrict__ W0, const float* __restrict__ B0,
    const float* __restrict__ W1, const float* __restrict__ B1,
    const float* __restrict__ W2, const float* __restrict__ B2,
    float* __restrict__ out, int N0_, int N1_, int N2_)
{
  const int br = blockIdx.y;
  const float* W; const float* bias; float* io; int n;
  if (br == 0)      { W = W0; bias = B0; io = out;                              n = N0_; }
  else if (br == 1) { W = W1; bias = B1; io = out + (size_t)N0_ * D_DIM;        n = N1_; }
  else              { W = W2; bias = B2; io = out + (size_t)(N0_ + N1_) * D_DIM; n = N2_; }

  const int mblk = blockIdx.x * BM;
  if (mblk >= n) return;

  const int tid  = threadIdx.x;
  const int lane = tid & 31;
  const int wave = tid >> 5;
  const int l16  = lane & 15;
  const int half = lane >> 4;

  v8f zero = {0.f, 0.f, 0.f, 0.f, 0.f, 0.f, 0.f, 0.f};
  v8f acc[8];
  #pragma unroll
  for (int t = 0; t < 8; ++t) acc[t] = zero;

  const float* aggRow = io + (size_t)(mblk + wave * 16 + l16) * D_DIM;

  #pragma unroll
  for (int kc = 0; kc < 4; ++kc) {         // K = 128 in chunks of 32
    const int kb = kc * KB;
    const int s  = half * 8;               // A lane K groups {0..7,16..23} / {8..15,24..31}
    v8u h0, l0, h1, l1;
    cvt8(aggRow + kb + s,      h0, l0);
    cvt8(aggRow + kb + s + 16, h1, l1);
    v16bf Ahi = combine16(h0, h1);
    v16bf Alo = combine16(l0, l1);

    #pragma unroll
    for (int t = 0; t < 8; ++t) {
      // B[k][n] = W[n][k]: W row-major is already the B (column-of-output) layout
      const float* wrow = W + (size_t)(t * 16 + l16) * D_DIM + kb + half * 8;
      v8u bh0, bl0, bh1, bl1;
      cvt8(wrow,     bh0, bl0);
      cvt8(wrow + 16, bh1, bl1);
      v16bf Bhi = combine16(bh0, bh1);
      v16bf Blo = combine16(bl0, bl1);
      acc[t] = WMMA_BF16(Ahi, Bhi, acc[t]);
      acc[t] = WMMA_BF16(Ahi, Blo, acc[t]);
      acc[t] = WMMA_BF16(Alo, Bhi, acc[t]);
    }
  }

  // epilogue: + bias, relu, store (overwrites this wave's own rows only)
  #pragma unroll
  for (int t = 0; t < 8; ++t) {
    const int nn = t * 16 + l16;
    const float bn = bias[nn];
    #pragma unroll
    for (int v = 0; v < 8; ++v) {
      const int m = mblk + wave * 16 + v + 8 * half;
      io[(size_t)m * D_DIM + nn] = fmaxf(acc[t][v] + bn, 0.0f);
    }
  }
}

// ---------------------------------------------------------------------------
// Host launcher.
// Inputs: 0:x0 1:x1 2:x2 3:L0 4:L1 5:L2 6:W0 7:b0 8:W1 9:b1 10:W2 11:b2
// Output: concat(relu-branch0, branch1, branch2), f32.
// ---------------------------------------------------------------------------
extern "C" void kernel_launch(void* const* d_in, const int* in_sizes, int n_in,
                              void* d_out, int out_size, void* d_ws, size_t ws_size,
                              hipStream_t stream) {
  (void)n_in; (void)out_size; (void)d_ws; (void)ws_size;
  const float* x0 = (const float*)d_in[0];
  const float* x1 = (const float*)d_in[1];
  const float* x2 = (const float*)d_in[2];
  const float* L0 = (const float*)d_in[3];
  const float* L1 = (const float*)d_in[4];
  const float* L2 = (const float*)d_in[5];
  const float* W0 = (const float*)d_in[6];
  const float* b0 = (const float*)d_in[7];
  const float* W1 = (const float*)d_in[8];
  const float* b1 = (const float*)d_in[9];
  const float* W2 = (const float*)d_in[10];
  const float* b2 = (const float*)d_in[11];
  float* out = (float*)d_out;

  const int N0 = in_sizes[0] / D_DIM;
  const int N1 = in_sizes[1] / D_DIM;
  const int N2 = in_sizes[2] / D_DIM;
  int nmax = N0 > N1 ? N0 : N1;
  if (N2 > nmax) nmax = N2;

  dim3 grid((unsigned)((nmax + BM - 1) / BM), 3, 1);
  dim3 block(256, 1, 1);

  scn_stage1<<<grid, block, 0, stream>>>(x0, x1, x2, L0, L1, L2, out, N0, N1, N2);
  scn_stage2<<<grid, block, 0, stream>>>(W0, b0, W1, b1, W2, b2, out, N0, N1, N2);
}
